// SearchTransfer_8091718385737
// MI455X (gfx1250) — matrix operand order repaired and checked by reading, not compiled, not run on hardware
//
#include <hip/hip_runtime.h>
#include <hip/hip_bf16.h>

// ---------------------------------------------------------------------------
// TTSR SearchTransfer for MI455X (gfx1250, wave32, WMMA bf16)
//
// Inputs (setup_inputs order):
//  0 lrsr_lv1 (unused)  1 lrsr_lv2 (unused)  2 lrsr_lv3 (1,256,80,80)
//  3 refsr_lv3 (1,256,80,80)  4 ref_lv1 (1,64,320,320)
//  5 ref_lv2 (1,128,160,160)  6 ref_lv3 (1,256,80,80)  7 return_attention(=0)
// Output: S(6400) | T_lv3(1638400) | T_lv2(3276800) | T_lv1(6553600), f32
// ---------------------------------------------------------------------------

typedef __attribute__((ext_vector_type(16))) __bf16 v16bf;
typedef __attribute__((ext_vector_type(8)))  float  v8f;

union BF16Frag {
    uint4 u[2];
    v16bf v;
};

__device__ __forceinline__ unsigned short f2bf(float f) {
    unsigned int u = __float_as_uint(f);
    u += 0x7FFFu + ((u >> 16) & 1u);          // round-to-nearest-even
    return (unsigned short)(u >> 16);
}
__device__ __forceinline__ float bf2f(unsigned short h) {
    return __uint_as_float(((unsigned int)h) << 16);
}

// ---------------------------------------------------------------------------
// Kernel 1: unfold (k=3,p=1,s=1) of a (256,80,80) tensor into a row-major
// [6400 rows][2304 K] bf16 matrix.  k-index = c*9 + ky*3 + kx.
// ---------------------------------------------------------------------------
__global__ __launch_bounds__(256) void pack_unfold3_kernel(
        const float* __restrict__ in, unsigned short* __restrict__ out)
{
    int t = blockIdx.x * 256 + threadIdx.x;           // 6400*2304 = 14,745,600
    if (t >= 6400 * 2304) return;
    int k = t % 2304;
    int r = t / 2304;
    int c  = k / 9, off = k % 9;
    int ky = off / 3, kx = off % 3;
    int y  = r / 80, x = r % 80;
    int sy = y + ky - 1, sx = x + kx - 1;
    float v = 0.f;
    if (sy >= 0 && sy < 80 && sx >= 0 && sx < 80)
        v = in[(c * 80 + sy) * 80 + sx];
    out[t] = f2bf(v);
}

// ---------------------------------------------------------------------------
// Kernel 2: inverse L2 norm of each 2304-length vector (A rows then B rows,
// stored contiguously).  One block per vector.
// ---------------------------------------------------------------------------
__global__ __launch_bounds__(256) void invnorm_kernel(
        const unsigned short* __restrict__ M, float* __restrict__ invN)
{
    __shared__ float red[256];
    int vec = blockIdx.x;                              // 0..12799
    const unsigned short* p = M + (size_t)vec * 2304;
    float s = 0.f;
    #pragma unroll
    for (int j = 0; j < 9; ++j) {
        float x = bf2f(p[threadIdx.x + j * 256]);
        s += x * x;
    }
    red[threadIdx.x] = s;
    __syncthreads();
    for (int st = 128; st > 0; st >>= 1) {
        if (threadIdx.x < st) red[threadIdx.x] += red[threadIdx.x + st];
        __syncthreads();
    }
    if (threadIdx.x == 0) {
        float n = sqrtf(red[0]);
        invN[vec] = 1.0f / fmaxf(n, 1e-12f);
    }
}

// ---------------------------------------------------------------------------
// Kernel 3: correlation + row-max/argmax via v_wmma_f32_16x16x32_bf16.
// Block = 8 waves, owns one 16-wide l-tile. Wave w scans r in [w*800,w*800+800)
// as 25 pairs of 16-row tiles (2 accumulators in flight).
//
// A frag (16-bit A 16x32, ISA 7.12.2): lane<16 -> row=lane, K {k0..k0+7,
// k0+16..k0+23}; lane>=16 -> row=lane-16, K {k0+8..15, k0+24..31}.
// B frag (same K order as A): lane<16 -> col=lane, K {k0..k0+7, k0+16..k0+23};
// lane>=16 -> col=lane-16, K {k0+8..15, k0+24..31}.
// C/D: vgpr i, lane L -> D[row = i + 8*(L>=16)][col = L%16].
// ---------------------------------------------------------------------------
__global__ __launch_bounds__(256) void corr_argmax_kernel(
        const unsigned short* __restrict__ A,      // refsr rows  [6400][2304]
        const unsigned short* __restrict__ Bm,     // lrsr  cols  [6400][2304]
        const float* __restrict__ invRf,
        const float* __restrict__ invLr,
        int*   __restrict__ Rarg,
        float* __restrict__ Sout)
{
    const int K = 2304;
    const int lbase = blockIdx.x * 16;
    const int wave  = threadIdx.x >> 5;
    const int lane  = threadIdx.x & 31;
    const int lr16  = lane & 15;
    const int hi    = lane >> 4;

    const unsigned short* Bcol  = Bm + (size_t)(lbase + lr16) * K + hi * 8;
    const unsigned short* Abase = A  + (size_t)lr16 * K + hi * 8;

    float best = -3.402823e38f;
    int   bidx = 0;

    for (int pair = 0; pair < 25; ++pair) {
        const int rbase = wave * 800 + pair * 32;
        const unsigned short* Arow0 = Abase + (size_t)rbase * K;
        const unsigned short* Arow1 = Arow0 + (size_t)16 * K;
        v8f acc0 = {};
        v8f acc1 = {};
        for (int k0 = 0; k0 < K; k0 += 32) {
            BF16Frag b, a0, a1;
            b.u[0]  = *(const uint4*)(Bcol  + k0);
            b.u[1]  = *(const uint4*)(Bcol  + k0 + 16);
            a0.u[0] = *(const uint4*)(Arow0 + k0);
            a0.u[1] = *(const uint4*)(Arow0 + k0 + 16);
            a1.u[0] = *(const uint4*)(Arow1 + k0);
            a1.u[1] = *(const uint4*)(Arow1 + k0 + 16);
            acc0 = __builtin_amdgcn_wmma_f32_16x16x32_bf16(
                false, a0.v, false, b.v, (short)0, acc0, false, false);
            acc1 = __builtin_amdgcn_wmma_f32_16x16x32_bf16(
                false, a1.v, false, b.v, (short)0, acc1, false, false);
        }
        #pragma unroll
        for (int i = 0; i < 8; ++i) {
            int r = rbase + hi * 8 + i;
            float v = acc0[i] * invRf[r];
            if (v > best) { best = v; bidx = r; }
        }
        #pragma unroll
        for (int i = 0; i < 8; ++i) {
            int r = rbase + 16 + hi * 8 + i;
            float v = acc1[i] * invRf[r];
            if (v > best) { best = v; bidx = r; }
        }
    }

    // combine the two row-halves held by lane L and lane L^16
    float ob = __shfl_xor(best, 16, 32);
    int   oi = __shfl_xor(bidx, 16, 32);
    if (ob > best || (ob == best && oi < bidx)) { best = ob; bidx = oi; }

    __shared__ float sbest[8][16];
    __shared__ int   sidx[8][16];
    if (lane < 16) { sbest[wave][lane] = best; sidx[wave][lane] = bidx; }
    __syncthreads();
    if (threadIdx.x < 16) {
        float bb = sbest[0][threadIdx.x];
        int   bi = sidx[0][threadIdx.x];
        #pragma unroll
        for (int w = 1; w < 8; ++w) {
            float v  = sbest[w][threadIdx.x];
            int   ii = sidx[w][threadIdx.x];
            if (v > bb || (v == bb && ii < bi)) { bb = v; bi = ii; }
        }
        int l = lbase + threadIdx.x;
        Rarg[l] = bi;
        Sout[l] = bb * invLr[l];         // S = R_star
    }
}

// ---------------------------------------------------------------------------
// Kernel 4: transfer + fold as a direct gather.
// For output (c,y,x) at scale F (k=3F,p=F,s=F,H=80F), with qy=y/F, ry=y%F:
// exactly patches (oy=qy-a, ox=qx-b), a,b in {-1,0,1}, contribute
// ref[c, F*(argY+a)+ry, F*(argX+b)+rx] (0 if OOB); sum / 9.
// ---------------------------------------------------------------------------
template <int F>
__global__ __launch_bounds__(256) void transfer_fold_kernel(
        const float* __restrict__ ref,
        const int*   __restrict__ Rarg,
        float* __restrict__ out,
        int C)
{
    const int H = 80 * F;
    int e = blockIdx.x * 256 + threadIdx.x;
    if (e >= C * H * H) return;
    int x = e % H;
    int t = e / H;
    int y = t % H;
    int c = t / H;
    int qy = y / F, ry = y - qy * F;
    int qx = x / F, rx = x - qx * F;
    float sum = 0.f;
    #pragma unroll
    for (int a = -1; a <= 1; ++a) {
        int oy = qy - a;
        if (oy < 0 || oy >= 80) continue;
        #pragma unroll
        for (int b = -1; b <= 1; ++b) {
            int ox = qx - b;
            if (ox < 0 || ox >= 80) continue;
            int arg  = Rarg[oy * 80 + ox];
            int argY = arg / 80;
            int argX = arg - argY * 80;
            int ay = argY + a;
            int ax = argX + b;
            if (ay < 0 || ay >= 80 || ax < 0 || ax >= 80) continue;
            sum += ref[((size_t)c * H + (ay * F + ry)) * H + (ax * F + rx)];
        }
    }
    out[e] = sum * (1.0f / 9.0f);
}

// ---------------------------------------------------------------------------
extern "C" void kernel_launch(void* const* d_in, const int* in_sizes, int n_in,
                              void* d_out, int out_size, void* d_ws, size_t ws_size,
                              hipStream_t stream)
{
    const float* lrsr3  = (const float*)d_in[2];
    const float* refsr3 = (const float*)d_in[3];
    const float* ref1   = (const float*)d_in[4];
    const float* ref2   = (const float*)d_in[5];
    const float* ref3   = (const float*)d_in[6];
    float* out = (float*)d_out;

    // workspace layout
    const size_t MAT_ELEMS = (size_t)6400 * 2304;          // 14,745,600
    char* ws = (char*)d_ws;
    unsigned short* Amat = (unsigned short*)ws;                      // refsr bf16
    unsigned short* Bmat = Amat + MAT_ELEMS;                         // lrsr  bf16
    float* invN = (float*)(ws + 2 * MAT_ELEMS * sizeof(unsigned short)); // 12800
    int*   Rarg = (int*)((char*)invN + 12800 * sizeof(float));       // 6400

    const int packBlocks = (int)((MAT_ELEMS + 255) / 256);           // 57600
    pack_unfold3_kernel<<<packBlocks, 256, 0, stream>>>(refsr3, Amat);
    pack_unfold3_kernel<<<packBlocks, 256, 0, stream>>>(lrsr3,  Bmat);

    invnorm_kernel<<<12800, 256, 0, stream>>>(Amat, invN);           // A then B

    corr_argmax_kernel<<<400, 256, 0, stream>>>(
        Amat, Bmat, invN, invN + 6400, Rarg, out /* S at offset 0 */);

    float* outT3 = out + 6400;
    float* outT2 = outT3 + 256 * 80 * 80;      // +1,638,400
    float* outT1 = outT2 + 128 * 160 * 160;    // +3,276,800

    transfer_fold_kernel<1><<< (256 * 80 * 80)   / 256, 256, 0, stream>>>(ref3, Rarg, outT3, 256);
    transfer_fold_kernel<2><<< (128 * 160 * 160) / 256, 256, 0, stream>>>(ref2, Rarg, outT2, 128);
    transfer_fold_kernel<4><<< (64 * 320 * 320)  / 256, 256, 0, stream>>>(ref1, Rarg, outT1, 64);
}
